// MultiHeadAttention_16595753632398
// MI455X (gfx1250) — matrix operand order repaired and checked by reading, not compiled, not run on hardware
//
#include <hip/hip_runtime.h>
#include <cstdint>
#include <cstddef>

#define EMBED  1024
#define HEADS  16
#define HD     64
#define NBATCH 4
#define SEQ    2048

typedef __attribute__((ext_vector_type(16))) _Float16     v16h;
typedef __attribute__((ext_vector_type(8)))  float        v8f;
typedef __attribute__((ext_vector_type(4)))  unsigned int u32x4;
typedef __attribute__((ext_vector_type(8)))  int          i32x8;
typedef __attribute__((ext_vector_type(4)))  int          i32x4;

union FragA {
    v16h     h;
    uint32_t u[8];
    _Float16 f[16];
};

// ---------------------------------------------------------------------------
// Tensor Data Mover: DMA one 2-D f32 tile (tile_w x tile_h elements, global
// row stride row_stride elements) into LDS at lds_byte_addr, inserting a
// 4-DWORD pad after every 64 DWORDs (-> 68-float LDS row stride).
// D# packing per CDNA5 ISA.
// ---------------------------------------------------------------------------
static __device__ __forceinline__ void tdm_load_tile_f32(
    uint32_t lds_byte_addr, const float* gptr,
    unsigned tile_w, unsigned tile_h, unsigned row_stride)
{
    const uint64_t ga = (uint64_t)(uintptr_t)gptr;

    u32x4 g0;
    g0[0] = 1u;                                            // count=1, user desc
    g0[1] = lds_byte_addr;                                 // lds_addr [63:32]
    g0[2] = (unsigned)(ga & 0xffffffffu);                  // global_addr lo
    g0[3] = (unsigned)((ga >> 32) & 0x01ffffffu)           // global_addr hi (57b)
            | (2u << 30);                                  // type=2 (image)

    const unsigned td1 = 1u << 20;                         // tensor_dim1 (large)
    i32x8 g1;
    g1[0] = (int)((2u << 16)        // data_size = 4 bytes
                | (1u << 20)        // pad_enable
                | (5u << 22)        // pad_interval: every 64 DWORDs
                | (3u << 25));      // pad_amount: 4 DWORDs
    g1[1] = (int)((row_stride & 0xffffu) << 16);           // tensor_dim0 lo16
    g1[2] = (int)((row_stride >> 16) | ((td1 & 0xffffu) << 16));
    g1[3] = (int)(((td1 >> 16) & 0xffffu) | (tile_w << 16));   // tile_dim0
    g1[4] = (int)(tile_h & 0xffffu);                       // tile_dim1, tile_dim2=0
    g1[5] = (int)row_stride;                               // tensor_dim0_stride
    g1[6] = 0;
    g1[7] = 0;

    const i32x4 z4 = {0, 0, 0, 0};
#if __clang_major__ >= 23
    const i32x8 z8 = {0, 0, 0, 0, 0, 0, 0, 0};
    __builtin_amdgcn_tensor_load_to_lds(g0, g1, z4, z4, z8, 0);
#else
    __builtin_amdgcn_tensor_load_to_lds(g0, g1, z4, z4, 0);
#endif
}

// ---------------------------------------------------------------------------
// Kernel 0: convert W_out (f32, [E][E] row-major, out = in @ W^T) to f16
// ---------------------------------------------------------------------------
__global__ void cvt_w_kernel(const float* __restrict__ W, _Float16* __restrict__ Wh) {
    int i = blockIdx.x * blockDim.x + threadIdx.x;
    Wh[i] = (_Float16)W[i];
}

// ---------------------------------------------------------------------------
// Kernel 1: flash-attention forward.  One block = (n, h, 128-query slab),
// 8 waves, each wave owns a 16-query x 64-dim output tile.
// K/V tiles are DMA'd into double-buffered LDS by the TDM (wave 0 issues,
// TENSORcnt + barrier for handoff); WMMAs run out of the other buffer.
// ---------------------------------------------------------------------------
__global__ __launch_bounds__(256, 1) void attn_fwd_kernel(
    const float* __restrict__ Q, const float* __restrict__ K,
    const float* __restrict__ V, const int* __restrict__ Mask,
    _Float16* __restrict__ attn16)
{
    __shared__ __align__(16) float    KsBuf[2][32][68];   // key tile (f32, TDM-padded rows)
    __shared__ __align__(16) float    VsBuf[2][32][68];   // value tile (f32)
    __shared__ __align__(16) _Float16 Ps[8][16][36];      // per-wave P tile (C->A bounce)

    const int tid  = threadIdx.x;
    const int wave = tid >> 5;
    const int lane = tid & 31;
    const int hi   = lane >> 4;          // 16-lane half
    const int ln   = lane & 15;

    const int qslabs = SEQ / 128;        // 16
    const int nh = blockIdx.x / qslabs;
    const int qb = blockIdx.x % qslabs;
    const int n  = nh / HEADS;
    const int h  = nh % HEADS;

    const int qrow0 = qb * 128 + wave * 16;

    const float* Kbase = K + ((size_t)n * SEQ) * EMBED + h * HD;
    const float* Vbase = V + ((size_t)n * SEQ) * EMBED + h * HD;

    // ---- Q tile as two A-fragments (head-dim 0-31 / 32-63), f32 -> f16 ----
    FragA qa[2];
    {
        const float* qp = Q + ((size_t)(n * SEQ + qrow0 + ln)) * EMBED + h * HD;
#pragma unroll
        for (int vv = 0; vv < 8; ++vv) {
            const int kk = 2 * vv + ((vv >= 4) ? 8 : 0) + hi * 8;   // ISA A-layout
            qa[0].f[2 * vv]     = (_Float16)qp[kk];
            qa[0].f[2 * vv + 1] = (_Float16)qp[kk + 1];
            qa[1].f[2 * vv]     = (_Float16)qp[32 + kk];
            qa[1].f[2 * vv + 1] = (_Float16)qp[32 + kk + 1];
        }
    }

    float mrow[8], lrow[8];
#pragma unroll
    for (int r = 0; r < 8; ++r) { mrow[r] = -3.0e38f; lrow[r] = 0.0f; }
    v8f oacc[4] = {};

    const float scale = 0.03125f;        // 1/sqrt(EMBED)
    const int* mp = Mask + (size_t)n * SEQ * SEQ;
    const int NCHUNK = SEQ / 32;         // 64

    // prologue: DMA chunk 0 into buffer 0 (wave 0 only -> one TDM op per tile)
    if (wave == 0) {
        tdm_load_tile_f32((uint32_t)(uintptr_t)&KsBuf[0][0][0], Kbase, HD, 32, EMBED);
        tdm_load_tile_f32((uint32_t)(uintptr_t)&VsBuf[0][0][0], Vbase, HD, 32, EMBED);
    }

    for (int kc = 0; kc < NCHUNK; ++kc) {
        const int kbase = kc * 32;
        const int cur   = kc & 1;

        __builtin_amdgcn_s_wait_tensorcnt(0);   // wave 0: tile for `cur` landed
        __syncthreads();                        // all waves: data ready, prev buffer free

        if (wave == 0 && kc + 1 < NCHUNK) {     // DMA next chunk into other buffer
            tdm_load_tile_f32((uint32_t)(uintptr_t)&KsBuf[cur ^ 1][0][0],
                              Kbase + (size_t)(kbase + 32) * EMBED, HD, 32, EMBED);
            tdm_load_tile_f32((uint32_t)(uintptr_t)&VsBuf[cur ^ 1][0][0],
                              Vbase + (size_t)(kbase + 32) * EMBED, HD, 32, EMBED);
        }

        const float (*Ksb)[68] = KsBuf[cur];
        const float (*Vsb)[68] = VsBuf[cur];

        // ---- scores: Q(16x64) x K^T(64x32) -> two 16x16 C tiles ----
        v8f s[2];
#pragma unroll
        for (int t = 0; t < 2; ++t) {
            v8f acc = {};
#pragma unroll
            for (int hc = 0; hc < 2; ++hc) {
                FragA kb;
#pragma unroll
                for (int vv = 0; vv < 8; ++vv) {
                    const float2 w = *(const float2*)&Ksb[t * 16 + ln][hc * 32 + 2 * vv + ((vv >= 4) ? 8 : 0) + hi * 8];
                    kb.f[2 * vv]     = (_Float16)w.x;
                    kb.f[2 * vv + 1] = (_Float16)w.y;
                }
                acc = __builtin_amdgcn_wmma_f32_16x16x32_f16(
                          false, qa[hc].h, false, kb.h, (short)0, acc, false, false);
            }
            s[t] = acc;
        }

        // ---- mask (pre-scale, as in reference) then scale ----
#pragma unroll
        for (int t = 0; t < 2; ++t) {
#pragma unroll
            for (int r = 0; r < 8; ++r) {
                const int qr   = qrow0 + hi * 8 + r;
                const int kcol = kbase + t * 16 + ln;
                const int mv   = mp[(size_t)qr * SEQ + kcol];
                s[t][r] = (mv != 0) ? s[t][r] * scale : -3.125e18f;  // -1e20/32
            }
        }

        // ---- online softmax: row stats replicated across each 16-lane half ----
        float alpha[8];
#pragma unroll
        for (int r = 0; r < 8; ++r) {
            float mx = fmaxf(s[0][r], s[1][r]);
#pragma unroll
            for (int off = 8; off >= 1; off >>= 1)
                mx = fmaxf(mx, __shfl_xor(mx, off, 32));
            const float mn = fmaxf(mrow[r], mx);
            alpha[r] = __expf(mrow[r] - mn);
            mrow[r]  = mn;
            const float p0 = __expf(s[0][r] - mn);
            const float p1 = __expf(s[1][r] - mn);
            s[0][r] = p0; s[1][r] = p1;
            float rs = p0 + p1;
#pragma unroll
            for (int off = 8; off >= 1; off >>= 1)
                rs += __shfl_xor(rs, off, 32);
            lrow[r] = lrow[r] * alpha[r] + rs;
        }

        // ---- rescale running output ----
#pragma unroll
        for (int t = 0; t < 4; ++t)
#pragma unroll
            for (int r = 0; r < 8; ++r)
                oacc[t][r] *= alpha[r];

        // ---- P: C-layout -> LDS -> A-layout fragment ----
#pragma unroll
        for (int r = 0; r < 8; ++r) {
            Ps[wave][hi * 8 + r][ln]      = (_Float16)s[0][r];
            Ps[wave][hi * 8 + r][16 + ln] = (_Float16)s[1][r];
        }
        __builtin_amdgcn_wave_barrier();
        FragA pa;
#pragma unroll
        for (int vv = 0; vv < 8; ++vv) {
            const int kk = 2 * vv + ((vv >= 4) ? 8 : 0) + hi * 8;
            pa.u[vv] = *(const uint32_t*)&Ps[wave][ln][kk];
        }

        // ---- O += P(16x32) x V(32x64) : 4 WMMAs ----
#pragma unroll
        for (int t = 0; t < 4; ++t) {
            FragA vb;
#pragma unroll
            for (int vv = 0; vv < 8; ++vv) {
                const int kk = 2 * vv + ((vv >= 4) ? 8 : 0) + hi * 8;   // ISA A-layout (table order)
                vb.f[2 * vv]     = (_Float16)Vsb[kk][t * 16 + ln];
                vb.f[2 * vv + 1] = (_Float16)Vsb[kk + 1][t * 16 + ln];
            }
            oacc[t] = __builtin_amdgcn_wmma_f32_16x16x32_f16(
                          false, pa.h, false, vb.h, (short)0, oacc[t], false, false);
        }
    }

    // ---- normalize and write f16 attention output (N,S,EMBED) ----
#pragma unroll
    for (int r = 0; r < 8; ++r) {
        const float inv = 1.0f / lrow[r];
        const int qr = qrow0 + hi * 8 + r;
        _Float16* op = attn16 + ((size_t)(n * SEQ + qr)) * EMBED + h * HD;
#pragma unroll
        for (int t = 0; t < 4; ++t)
            op[t * 16 + ln] = (_Float16)(oacc[t][r] * inv);
    }
}

// ---------------------------------------------------------------------------
// Kernel 2: out = attn(8192x1024,f16) @ W^T(f16) + b, f32 out.
// One block per 16-row M tile; each wave handles 8 N-tiles, sharing the
// A-fragment per K-chunk.
// ---------------------------------------------------------------------------
__global__ __launch_bounds__(256, 1) void out_proj_kernel(
    const _Float16* __restrict__ A, const _Float16* __restrict__ Wh,
    const float* __restrict__ bias, float* __restrict__ out)
{
    const int tid  = threadIdx.x;
    const int wave = tid >> 5;
    const int lane = tid & 31;
    const int hi   = lane >> 4;
    const int ln   = lane & 15;
    const int mtile = blockIdx.x;                 // 512 tiles of 16 rows

    v8f acc[8] = {};
    const _Float16* ap = A + ((size_t)(mtile * 16 + ln)) * EMBED;

    for (int kc = 0; kc < EMBED / 32; ++kc) {
        FragA a;
#pragma unroll
        for (int vv = 0; vv < 8; ++vv) {
            const int kk = 2 * vv + ((vv >= 4) ? 8 : 0) + hi * 8;
            a.u[vv] = *(const uint32_t*)(ap + kc * 32 + kk);
        }
#pragma unroll
        for (int j = 0; j < 8; ++j) {
            const int ntile = wave * 8 + j;
            const _Float16* wp = Wh + ((size_t)(ntile * 16 + ln)) * EMBED + kc * 32;
            FragA b;
#pragma unroll
            for (int vv = 0; vv < 8; ++vv)
                b.u[vv] = *(const uint32_t*)(wp + 2 * vv + ((vv >= 4) ? 8 : 0) + hi * 8);
            acc[j] = __builtin_amdgcn_wmma_f32_16x16x32_f16(
                         false, a.h, false, b.h, (short)0, acc[j], false, false);
        }
    }

#pragma unroll
    for (int j = 0; j < 8; ++j) {
        const int ntile = wave * 8 + j;
        const int col = ntile * 16 + ln;
        const float bb = bias[col];
#pragma unroll
        for (int r = 0; r < 8; ++r) {
            const int row = mtile * 16 + hi * 8 + r;
            out[(size_t)row * EMBED + col] = acc[j][r] + bb;
        }
    }
}

// ---------------------------------------------------------------------------
extern "C" void kernel_launch(void* const* d_in, const int* in_sizes, int n_in,
                              void* d_out, int out_size, void* d_ws, size_t ws_size,
                              hipStream_t stream) {
    const float* V  = (const float*)d_in[0];
    const float* K  = (const float*)d_in[1];
    const float* Q  = (const float*)d_in[2];
    const int*   M  = (const int*)d_in[3];
    const float* W  = (const float*)d_in[4];
    const float* b  = (const float*)d_in[5];
    float* out = (float*)d_out;

    // workspace: [0, 2MB) W in f16 ; [2MB, 18MB) attention output in f16
    _Float16* Wh     = (_Float16*)d_ws;
    _Float16* attn16 = Wh + (size_t)EMBED * EMBED;

    cvt_w_kernel<<<(EMBED * EMBED) / 256, 256, 0, stream>>>(W, Wh);
    attn_fwd_kernel<<<NBATCH * HEADS * (SEQ / 128), 256, 0, stream>>>(Q, K, V, M, attn16);
    out_proj_kernel<<<(NBATCH * SEQ) / 16, 256, 0, stream>>>(attn16, Wh, b, out);
}
